// BERT_RAT_89953795047932
// MI455X (gfx1250) — matrix operand order repaired and checked by reading, not compiled, not run on hardware
//
#include <hip/hip_runtime.h>
#include <math.h>

// ---------------- model constants ----------------
static constexpr int Bc   = 4;
static constexpr int Sc   = 512;
static constexpr int Lc   = 12;
static constexpr int Hc   = 12;
static constexpr int Dc   = 768;
static constexpr int DHc  = 64;
static constexpr int DFFc = 3072;
static constexpr int Rc   = 50;
static constexpr int RATL = 6;

typedef __attribute__((ext_vector_type(16))) _Float16 v16h;
typedef __attribute__((ext_vector_type(2)))  __fp16   v2hp;  // cvt_pkrtz result type
typedef __attribute__((ext_vector_type(8)))  float    v8f;

// =====================================================================
// Batched GEMM: C = A[M,K] * B[K,N] (+bias)(+Res)(gelu?)
//   f32 global -> packed f16 (v_cvt_pk_rtz) -> fragment-swizzled LDS
//   -> ds_load_b128 fragment reads -> v_wmma_f32_16x16x32_f16.
// Block: 256 threads = 8 wave32 waves. Block tile 128(M) x 64(N), K-step 32.
// Wave (w&3) owns 32 M-rows, (w>>2) owns 32 N-cols -> 4 accumulators.
// Double-buffered LDS: stage tile k+1 while WMMAs of tile k execute;
// one workgroup barrier per K-step. All staging addresses (global ptrs
// and LDS offsets) are precomputed per-thread; loop body only advances
// pointers by a constant stride.
//
// LDS fragment layout (this kernel's contiguous k order, same for A and B):
//   A frag: lane=(m&15)+16*(k>>4),     elem=k&15
//   B frag: lane=(n&15)+16*(k>>4),     elem=k&15
// stored [tile][lane][16 halves] -> one fragment = 32 contiguous bytes
// per lane (two ds_load_b128).
// =====================================================================
__global__ __launch_bounds__(256) void gemm_wmma_kernel(
    const float* __restrict__ A, long lda, long sAb, long sAh,
    const float* __restrict__ Bm, long ldb, long sBb, long sBh, int transB,
    const float* __restrict__ bias,
    const float* __restrict__ Res, long ldr, long sRb, long sRh,
    float* __restrict__ C, long ldc, long sCb, long sCh,
    int K, int Hb, int gelu)
{
  const int z  = blockIdx.z;
  const int zb = z / Hb;
  const int zh = z % Hb;
  A  += zb * sAb + zh * sAh;
  Bm += zb * sBb + zh * sBh;
  C  += zb * sCb + zh * sCh;
  if (Res) Res += zb * sRb + zh * sRh;

  const int m0 = blockIdx.y * 128;
  const int n0 = blockIdx.x * 64;

  __shared__ _Float16 Af[2][8 * 32 * 16];   // ping-pong, 2 x 8 KB
  __shared__ _Float16 Bf[2][4 * 32 * 16];   // ping-pong, 2 x 4 KB

  const int tid  = threadIdx.x;
  const int lane = tid & 31;
  const int w    = tid >> 5;
  const int wm   = (w & 3) * 32;
  const int wn   = (w >> 2) * 32;
  const int lh   = lane >> 4;
  const int ll   = lane & 15;
  const int mt0  = (w & 3) * 2;
  const int nt0  = (w >> 2) * 2;

  // ---- precompute A staging: 8 chunks, pair col fixed per thread ----
  const int kA  = 2 * (tid & 15);
  const int lnA = (tid >> 4) + 16 * (kA >> 4);
  const int e0A = kA & 15;
  const float* aPtr[8];
  int aLds[8];
  #pragma unroll
  for (int i = 0; i < 8; ++i) {
    const int row = (tid >> 4) + 16 * i;           // mt == i
    aPtr[i] = A + (long)(m0 + row) * lda + kA;
    aLds[i] = ((i * 32 + lnA) << 4) + e0A;
  }

  // ---- precompute B staging: 4 chunks ----
  const float* bPtr[4];
  int bLds[4], bLds2[4];
  long bAdv;
  if (!transB) {
    const int c = 2 * (tid & 31);                  // n col, even
    #pragma unroll
    for (int i = 0; i < 4; ++i) {
      const int r = (tid >> 5) + 8 * i;            // k row 0..31
      bPtr[i] = Bm + (long)r * ldb + (n0 + c);
      const int e   = r & 15;
      const int nt  = c >> 4;
      const int ln0 = (c & 15) + 16 * (r >> 4);
      bLds[i]  = ((nt * 32 + ln0) << 4) + e;
      bLds2[i] = ((nt * 32 + ln0 + 1) << 4) + e;
    }
    bAdv = 32 * ldb;
  } else {
    const int r   = 2 * (tid & 15);                // k row, even
    const int e   = r & 15;
    const int lnB = (tid >> 4) + 16 * (r >> 4);
    #pragma unroll
    for (int i = 0; i < 4; ++i) {
      const int c = (tid >> 4) + 16 * i;           // nt == i
      bPtr[i] = Bm + (long)(n0 + c) * ldb + r;
      bLds[i] = ((i * 32 + lnB) << 4) + e;
      bLds2[i] = 0;
    }
    bAdv = 32;
  }

  // ---- two-phase stager: batch all global loads, then convert+store ----
  auto stage = [&](int buf) {
    float2 va[8], vb[4];
    #pragma unroll
    for (int i = 0; i < 8; ++i) va[i] = *(const float2*)aPtr[i];
    #pragma unroll
    for (int i = 0; i < 4; ++i) vb[i] = *(const float2*)bPtr[i];
    _Float16* Ad = Af[buf];
    _Float16* Bd = Bf[buf];
    #pragma unroll
    for (int i = 0; i < 8; ++i) {
      __builtin_prefetch(aPtr[i] + 32, 0, 3);
      aPtr[i] += 32;
      *(v2hp*)&Ad[aLds[i]] = __builtin_amdgcn_cvt_pkrtz(va[i].x, va[i].y);
    }
    if (!transB) {
      #pragma unroll
      for (int i = 0; i < 4; ++i) {
        __builtin_prefetch(bPtr[i] + bAdv, 0, 3);
        bPtr[i] += bAdv;
        Bd[bLds[i]]  = (_Float16)vb[i].x;
        Bd[bLds2[i]] = (_Float16)vb[i].y;
      }
    } else {
      #pragma unroll
      for (int i = 0; i < 4; ++i) {
        __builtin_prefetch(bPtr[i] + bAdv, 0, 3);
        bPtr[i] += bAdv;
        *(v2hp*)&Bd[bLds[i]] = __builtin_amdgcn_cvt_pkrtz(vb[i].x, vb[i].y);
      }
    }
  };

  v8f acc[2][2] = {{{}, {}}, {{}, {}}};

  const int nk = K >> 5;
  stage(0);
  __syncthreads();

  for (int kt = 0; kt < nk; ++kt) {
    const int cur = kt & 1;
    const _Float16* Ac = Af[cur];
    const _Float16* Bc = Bf[cur];

    // fragment reads: 32 contiguous bytes/lane -> 2x ds_load_b128 each
    v16h a0 = *(const v16h*)&Ac[((mt0 * 32 + lane) << 4)];
    v16h a1 = *(const v16h*)&Ac[(((mt0 + 1) * 32 + lane) << 4)];
    v16h b0 = *(const v16h*)&Bc[((nt0 * 32 + lane) << 4)];
    v16h b1 = *(const v16h*)&Bc[(((nt0 + 1) * 32 + lane) << 4)];

    // overlap: stage next tile into the other buffer while WMMAs run
    if (kt + 1 < nk) stage((kt + 1) & 1);

    acc[0][0] = __builtin_amdgcn_wmma_f32_16x16x32_f16(false, a0, false, b0,
                                                       (short)0, acc[0][0], false, false);
    acc[0][1] = __builtin_amdgcn_wmma_f32_16x16x32_f16(false, a0, false, b1,
                                                       (short)0, acc[0][1], false, false);
    acc[1][0] = __builtin_amdgcn_wmma_f32_16x16x32_f16(false, a1, false, b0,
                                                       (short)0, acc[1][0], false, false);
    acc[1][1] = __builtin_amdgcn_wmma_f32_16x16x32_f16(false, a1, false, b1,
                                                       (short)0, acc[1][1], false, false);
    __syncthreads();
  }

  // ---- epilogue: C/D layout (lane ll = column, VGPR j = row j+8*lh) ----
  #pragma unroll
  for (int mi = 0; mi < 2; ++mi) {
    #pragma unroll
    for (int ni = 0; ni < 2; ++ni) {
      const int nn = n0 + wn + ni * 16 + ll;
      const float bn = bias ? bias[nn] : 0.0f;
      #pragma unroll
      for (int j = 0; j < 8; ++j) {
        const int mm = m0 + wm + mi * 16 + j + 8 * lh;
        float v = acc[mi][ni][j] + bn;
        if (Res)  v += Res[(long)mm * ldr + nn];
        if (gelu) v = 0.5f * v * (1.0f + erff(v * 0.70710678118654752f));
        C[(long)mm * ldc + nn] = v;
      }
    }
  }
}

// =====================================================================
// Embeddings + LayerNorm: one block per token, 768 = 256*3
// =====================================================================
__global__ __launch_bounds__(256) void embed_ln_kernel(
    const int* __restrict__ ids,
    const float* __restrict__ we, const float* __restrict__ pe,
    const float* __restrict__ te,
    const float* __restrict__ g, const float* __restrict__ bb,
    float* __restrict__ h)
{
  const int tok = blockIdx.x;
  const int s   = tok % Sc;
  const int id  = ids[tok];
  const int tid = threadIdx.x;

  float v0 = we[(long)id * Dc + tid]       + pe[(long)s * Dc + tid]       + te[tid];
  float v1 = we[(long)id * Dc + tid + 256] + pe[(long)s * Dc + tid + 256] + te[tid + 256];
  float v2 = we[(long)id * Dc + tid + 512] + pe[(long)s * Dc + tid + 512] + te[tid + 512];

  __shared__ float red[256];
  red[tid] = v0 + v1 + v2;
  __syncthreads();
  for (int o = 128; o > 0; o >>= 1) { if (tid < o) red[tid] += red[tid + o]; __syncthreads(); }
  const float mean = red[0] * (1.0f / Dc);
  __syncthreads();

  const float d0 = v0 - mean, d1 = v1 - mean, d2 = v2 - mean;
  red[tid] = d0 * d0 + d1 * d1 + d2 * d2;
  __syncthreads();
  for (int o = 128; o > 0; o >>= 1) { if (tid < o) red[tid] += red[tid + o]; __syncthreads(); }
  const float rstd = rsqrtf(red[0] * (1.0f / Dc) + 1e-12f);

  float* hr = h + (long)tok * Dc;
  hr[tid]       = d0 * rstd * g[tid]       + bb[tid];
  hr[tid + 256] = d1 * rstd * g[tid + 256] + bb[tid + 256];
  hr[tid + 512] = d2 * rstd * g[tid + 512] + bb[tid + 512];
}

// =====================================================================
// LayerNorm over last dim (768): one block per row
// =====================================================================
__global__ __launch_bounds__(256) void layernorm_kernel(
    const float* __restrict__ x,
    const float* __restrict__ g, const float* __restrict__ bb,
    float* __restrict__ y)
{
  const long row = blockIdx.x;
  const int  tid = threadIdx.x;
  const float* xr = x + row * Dc;
  float v0 = xr[tid], v1 = xr[tid + 256], v2 = xr[tid + 512];

  __shared__ float red[256];
  red[tid] = v0 + v1 + v2;
  __syncthreads();
  for (int o = 128; o > 0; o >>= 1) { if (tid < o) red[tid] += red[tid + o]; __syncthreads(); }
  const float mean = red[0] * (1.0f / Dc);
  __syncthreads();

  const float d0 = v0 - mean, d1 = v1 - mean, d2 = v2 - mean;
  red[tid] = d0 * d0 + d1 * d1 + d2 * d2;
  __syncthreads();
  for (int o = 128; o > 0; o >>= 1) { if (tid < o) red[tid] += red[tid + o]; __syncthreads(); }
  const float rstd = rsqrtf(red[0] * (1.0f / Dc) + 1e-12f);

  float* yr = y + row * Dc;
  yr[tid]       = d0 * rstd * g[tid]       + bb[tid];
  yr[tid + 256] = d1 * rstd * g[tid + 256] + bb[tid + 256];
  yr[tid + 512] = d2 * rstd * g[tid + 512] + bb[tid + 512];
}

// =====================================================================
// Softmax over t (512) with 1/sqrt(DH) scale and additive mask.
// =====================================================================
__global__ __launch_bounds__(256) void softmax_kernel(
    float* __restrict__ sc, const int* __restrict__ am)
{
  const int row = blockIdx.x;               // ((b*H + h)*S + s)
  const int b   = row / (Hc * Sc);
  float* p = sc + (long)row * Sc;
  const int tid = threadIdx.x;

  const float mk0 = (1.0f - (float)am[b * Sc + tid])       * -10000.0f;
  const float mk1 = (1.0f - (float)am[b * Sc + tid + 256]) * -10000.0f;
  float x0 = p[tid] * 0.125f + mk0;         // 1/sqrt(64)
  float x1 = p[tid + 256] * 0.125f + mk1;

  __shared__ float red[256];
  red[tid] = fmaxf(x0, x1);
  __syncthreads();
  for (int o = 128; o > 0; o >>= 1) { if (tid < o) red[tid] = fmaxf(red[tid], red[tid + o]); __syncthreads(); }
  const float m = red[0];
  __syncthreads();

  const float e0 = __expf(x0 - m), e1 = __expf(x1 - m);
  red[tid] = e0 + e1;
  __syncthreads();
  for (int o = 128; o > 0; o >>= 1) { if (tid < o) red[tid] += red[tid + o]; __syncthreads(); }
  const float inv = 1.0f / red[0];
  p[tid]       = e0 * inv;
  p[tid + 256] = e1 * inv;
}

// =====================================================================
// RAT score add (layer 6): scores[b,h,s,t] += q[b,h,s,:] . key_rel[rel[b,s,t],:]
// =====================================================================
__global__ __launch_bounds__(256) void rat_scores_kernel(
    const float* __restrict__ q, const int* __restrict__ rel,
    const float* __restrict__ krl, float* __restrict__ scores)
{
  const int s = blockIdx.x;
  const int b = blockIdx.y;
  const int tid = threadIdx.x;

  __shared__ float qs[Dc];
  __shared__ float krs[Rc * DHc];
  for (int i = tid; i < Dc; i += 256) qs[i] = q[((long)b * Sc + s) * Dc + i];
  for (int i = tid; i < Rc * DHc; i += 256) krs[i] = krl[i];
  __syncthreads();

  const int* relrow = rel + ((long)b * Sc + s) * Sc;
  #pragma unroll
  for (int u = 0; u < 2; ++u) {
    const int t = tid + u * 256;
    const int r = relrow[t];
    const float* kr = &krs[r * DHc];
    float sum[Hc];
    #pragma unroll
    for (int h = 0; h < Hc; ++h) sum[h] = 0.0f;
    for (int d = 0; d < DHc; ++d) {
      const float kv = kr[d];
      #pragma unroll
      for (int h = 0; h < Hc; ++h) sum[h] += qs[h * DHc + d] * kv;
    }
    #pragma unroll
    for (int h = 0; h < Hc; ++h)
      scores[(((long)(b * Hc + h) * Sc) + s) * Sc + t] += sum[h];
  }
}

// =====================================================================
// RAT context add (layer 6): ctx[b,s,h*64+d] += sum_t probs*val_rel[rel,d]
// =====================================================================
__global__ __launch_bounds__(256) void rat_ctx_kernel(
    const float* __restrict__ probs, const int* __restrict__ rel,
    const float* __restrict__ vrl, float* __restrict__ ctx)
{
  const int s = blockIdx.x;
  const int b = blockIdx.y;
  const int tid = threadIdx.x;

  __shared__ float vrs[Rc * DHc];   // 12.8 KB
  __shared__ float ps[Hc * Sc];     // 24 KB
  __shared__ int   rs[Sc];          // 2 KB
  for (int i = tid; i < Rc * DHc; i += 256) vrs[i] = vrl[i];
  for (int i = tid; i < Sc; i += 256) rs[i] = rel[((long)b * Sc + s) * Sc + i];
  for (int i = tid; i < Hc * Sc; i += 256) {
    const int h = i >> 9, t = i & 511;
    ps[i] = probs[(((long)(b * Hc + h) * Sc) + s) * Sc + t];
  }
  __syncthreads();

  float acc[3] = {0.0f, 0.0f, 0.0f};
  for (int t = 0; t < Sc; ++t) {
    const int r = rs[t];
    #pragma unroll
    for (int u = 0; u < 3; ++u) {
      const int idx = tid + u * 256;
      const int h = idx >> 6;
      const int d = idx & 63;
      acc[u] += ps[(h << 9) + t] * vrs[r * DHc + d];
    }
  }
  #pragma unroll
  for (int u = 0; u < 3; ++u) {
    const int idx = tid + u * 256;
    ctx[((long)b * Sc + s) * Dc + idx] += acc[u];
  }
}

// =====================================================================
// Pooler: pooled[b,j] = tanh(h[b,0,:] @ pool_w[:,j] + pool_b[j])
// =====================================================================
__global__ __launch_bounds__(256) void pooler_kernel(
    const float* __restrict__ hfin, const float* __restrict__ pw,
    const float* __restrict__ pb, float* __restrict__ out)
{
  const int gid = blockIdx.x * 256 + threadIdx.x;
  if (gid >= Bc * Dc) return;
  const int b = gid / Dc;
  const int j = gid % Dc;
  const float* hr = hfin + (long)b * Sc * Dc;  // s = 0 row
  float sum = pb[j];
  for (int d = 0; d < Dc; ++d) sum += hr[d] * pw[(long)d * Dc + j];
  out[gid] = tanhf(sum);
}

// =====================================================================
// Host orchestration
// =====================================================================
extern "C" void kernel_launch(void* const* d_in, const int* in_sizes, int n_in,
                              void* d_out, int out_size, void* d_ws, size_t ws_size,
                              hipStream_t stream)
{
  const int*   input_ids = (const int*)  d_in[0];
  const int*   relations = (const int*)  d_in[1];
  const int*   attn_mask = (const int*)  d_in[2];
  const float* word_emb  = (const float*)d_in[3];
  const float* pos_emb   = (const float*)d_in[4];
  const float* type_emb  = (const float*)d_in[5];
  const float* emb_ln_g  = (const float*)d_in[6];
  const float* emb_ln_b  = (const float*)d_in[7];
  const float* Wq        = (const float*)d_in[8];
  const float* bq        = (const float*)d_in[9];
  const float* Wk        = (const float*)d_in[10];
  const float* bk        = (const float*)d_in[11];
  const float* Wv        = (const float*)d_in[12];
  const float* bv        = (const float*)d_in[13];
  const float* Wo        = (const float*)d_in[14];
  const float* bo        = (const float*)d_in[15];
  const float* ln1_g     = (const float*)d_in[16];
  const float* ln1_b     = (const float*)d_in[17];
  const float* Wi        = (const float*)d_in[18];
  const float* bi        = (const float*)d_in[19];
  const float* Wf        = (const float*)d_in[20];
  const float* bf        = (const float*)d_in[21];
  const float* ln2_g     = (const float*)d_in[22];
  const float* ln2_b     = (const float*)d_in[23];
  const float* key_rel   = (const float*)d_in[24];
  const float* val_rel   = (const float*)d_in[25];
  const float* pool_w    = (const float*)d_in[26];
  const float* pool_b    = (const float*)d_in[27];

  const long BSD = (long)Bc * Sc * Dc;

  float* ws   = (float*)d_ws;
  float* hbuf = ws;                       // [B,S,D]
  float* xbuf = hbuf + BSD;               // pre-LN residual sums
  float* abuf = xbuf + BSD;               // attn_out
  float* qb   = abuf + BSD;
  float* kb   = qb   + BSD;
  float* vb   = kb   + BSD;
  float* ctx  = vb   + BSD;
  float* ffn  = ctx  + BSD;               // [B,S,DFF]
  float* sc   = ffn  + (long)Bc * Sc * DFFc;   // [B,H,S,S]

  float* out_h    = (float*)d_out;        // final hidden states
  float* out_pool = out_h + BSD;          // pooled

  auto gemm = [&](const float* A, long lda, long sAb, long sAh,
                  const float* Bm, long ldb, long sBb, long sBh, int transB,
                  const float* bias,
                  const float* Res, long ldr,
                  float* C, long ldc, long sCb, long sCh,
                  int M, int N, int K, int Hb, int nbatch, int gelu) {
    dim3 grid(N / 64, M / 128, nbatch);
    gemm_wmma_kernel<<<grid, 256, 0, stream>>>(
        A, lda, sAb, sAh, Bm, ldb, sBb, sBh, transB,
        bias, Res, ldr, 0L, 0L, C, ldc, sCb, sCh, K, Hb, gelu);
  };

  // ---- embeddings ----
  embed_ln_kernel<<<Bc * Sc, 256, 0, stream>>>(
      input_ids, word_emb, pos_emb, type_emb, emb_ln_g, emb_ln_b, hbuf);

  const long DD = (long)Dc * Dc;
  const long sBH_A = (long)Sc * Dc;   // per-batch stride in [B,S,D]
  const long sHH   = (long)DHc;       // per-head column offset
  const long sScB  = (long)Hc * Sc * Sc;
  const long sScH  = (long)Sc * Sc;

  for (int i = 0; i < Lc; ++i) {
    const float* Wq_i = Wq + (long)i * DD;   const float* bq_i = bq + (long)i * Dc;
    const float* Wk_i = Wk + (long)i * DD;   const float* bk_i = bk + (long)i * Dc;
    const float* Wv_i = Wv + (long)i * DD;   const float* bv_i = bv + (long)i * Dc;
    const float* Wo_i = Wo + (long)i * DD;   const float* bo_i = bo + (long)i * Dc;
    const float* Wi_i = Wi + (long)i * Dc * DFFc; const float* bi_i = bi + (long)i * DFFc;
    const float* Wf_i = Wf + (long)i * DFFc * Dc; const float* bf_i = bf + (long)i * Dc;

    // Q, K, V projections: [2048,768] @ [768,768] + bias
    gemm(hbuf, Dc, 0, 0, Wq_i, Dc, 0, 0, 0, bq_i, nullptr, 0,
         qb, Dc, 0, 0, Bc * Sc, Dc, Dc, 1, 1, 0);
    gemm(hbuf, Dc, 0, 0, Wk_i, Dc, 0, 0, 0, bk_i, nullptr, 0,
         kb, Dc, 0, 0, Bc * Sc, Dc, Dc, 1, 1, 0);
    gemm(hbuf, Dc, 0, 0, Wv_i, Dc, 0, 0, 0, bv_i, nullptr, 0,
         vb, Dc, 0, 0, Bc * Sc, Dc, Dc, 1, 1, 0);

    // scores = q @ k^T per (b,h): M=S, N=S, K=DH
    gemm(qb, Dc, sBH_A, sHH, kb, Dc, sBH_A, sHH, 1, nullptr, nullptr, 0,
         sc, Sc, sScB, sScH, Sc, Sc, DHc, Hc, Bc * Hc, 0);

    if (i == RATL) {
      rat_scores_kernel<<<dim3(Sc, Bc), 256, 0, stream>>>(
          qb, relations, key_rel + (long)RATL * Rc * DHc, sc);
    }

    softmax_kernel<<<Bc * Hc * Sc, 256, 0, stream>>>(sc, attn_mask);

    // ctx = probs @ v per (b,h): M=S, N=DH, K=S
    gemm(sc, Sc, sScB, sScH, vb, Dc, sBH_A, sHH, 0, nullptr, nullptr, 0,
         ctx, Dc, sBH_A, sHH, Sc, DHc, Sc, Hc, Bc * Hc, 0);

    if (i == RATL) {
      rat_ctx_kernel<<<dim3(Sc, Bc), 256, 0, stream>>>(
          sc, relations, val_rel + (long)RATL * Rc * DHc, ctx);
    }

    // O projection + bias + residual, then LN1 -> abuf
    gemm(ctx, Dc, 0, 0, Wo_i, Dc, 0, 0, 0, bo_i, hbuf, Dc,
         xbuf, Dc, 0, 0, Bc * Sc, Dc, Dc, 1, 1, 0);
    layernorm_kernel<<<Bc * Sc, 256, 0, stream>>>(xbuf, ln1_g + (long)i * Dc,
                                                  ln1_b + (long)i * Dc, abuf);

    // FFN up + exact GELU
    gemm(abuf, Dc, 0, 0, Wi_i, DFFc, 0, 0, 0, bi_i, nullptr, 0,
         ffn, DFFc, 0, 0, Bc * Sc, DFFc, Dc, 1, 1, 1);

    // FFN down + bias + residual, then LN2 -> hbuf (last layer -> d_out)
    gemm(ffn, DFFc, 0, 0, Wf_i, Dc, 0, 0, 0, bf_i, abuf, Dc,
         xbuf, Dc, 0, 0, Bc * Sc, Dc, DFFc, 1, 1, 0);
    float* lnout = (i == Lc - 1) ? out_h : hbuf;
    layernorm_kernel<<<Bc * Sc, 256, 0, stream>>>(xbuf, ln2_g + (long)i * Dc,
                                                  ln2_b + (long)i * Dc, lnout);
  }

  // ---- pooler ----
  pooler_kernel<<<(Bc * Dc + 255) / 256, 256, 0, stream>>>(
      out_h, pool_w, pool_b, out_pool);

  (void)in_sizes; (void)n_in; (void)out_size; (void)ws_size;
}
